// EetqLinear_84241488544414
// MI455X (gfx1250) — matrix operand order repaired and checked by reading, not compiled, not run on hardware
//
#include <hip/hip_runtime.h>
#include <hip/hip_bf16.h>
#include <stdint.h>

// ---------------------------------------------------------------------------
// w8a16 GEMM for MI455X (gfx1250, wave32, WMMA f32_16x16x32_f16)
//   out[M,N] = x[M,K] @ (int8 w[K,N] * scale[N]) + bias[N]
//   M = 8192, K = 4096, N = 16384
//
// BM=256 x BN=128 block, BK=64, 256 threads (8 waves), wave tile 128x32.
// A staged fp16 in LDS via GLOBAL_LOAD_ASYNC_TO_LDS_B128 (ASYNCcnt path);
// B dequantized int8->fp16 at fill time via v_perm byte gather + fp16
// magic-number fma, stored [N][K] so fragments are contiguous b128 LDS loads.
// ---------------------------------------------------------------------------

typedef __attribute__((ext_vector_type(16))) _Float16 v16h;
typedef __attribute__((ext_vector_type(8)))  float    v8f;
typedef __attribute__((ext_vector_type(2)))  _Float16 half2v;
typedef __attribute__((ext_vector_type(4)))  int      v4i;

#define GEMM_M 8192
#define GEMM_K 4096
#define GEMM_N 16384

#define BM 256
#define BN 128
#define BK 64
#define LDA (BK + 8)   // padded f16 stride: 144B rows, 16B aligned, bank-spread
#define LDB (BK + 8)

#define AS1 __attribute__((address_space(1)))
#define AS3 __attribute__((address_space(3)))

#if __has_builtin(__builtin_amdgcn_global_load_async_to_lds_b128)
#define HAVE_ASYNC_LDS 1
#else
#define HAVE_ASYNC_LDS 0
#endif

union AFrag { v16h h; uint4 q[2]; };

// dequant 2 int8 (byte j of qlo -> low half, byte j of qhi -> high half)
// h = 0x6400 | (u8 ^ 0x80)  ==  1152 + v  (exact in fp16)
// result = h * s + (-1152 * s) = v * s     (one v_pk_fma_f16)
__device__ __forceinline__ unsigned dq2(unsigned qhi, unsigned qlo, unsigned j,
                                        half2v s2, half2v c2) {
    unsigned u = __builtin_amdgcn_perm(qhi, qlo, 0x00040000u + j * 0x00010001u);
    u = (u & 0x00ff00ffu) ^ 0x64806480u;
    half2v h = __builtin_bit_cast(half2v, u);
    half2v r = h * s2 + c2;
    return __builtin_bit_cast(unsigned, r);
}

__global__ __launch_bounds__(256)
void w8a16_gemm_wmma(const _Float16* __restrict__ x,
                     const int8_t*   __restrict__ wq,
                     const _Float16* __restrict__ scales,
                     const _Float16* __restrict__ bias,
                     _Float16*       __restrict__ out) {
    __shared__ _Float16 Als[BM][LDA];   // [m][k]
    __shared__ _Float16 Bls[BN][LDB];   // dequantized weight, [n][k]

    const int tid  = threadIdx.x;
    const int lane = tid & 31;
    const int wid  = tid >> 5;          // 8 waves
    const int ln15 = lane & 15;
    const int lhi  = lane >> 4;

    const int gm0 = blockIdx.y * BM;
    const int gn0 = blockIdx.x * BN;

    // wave grid: 2 (M) x 4 (N); wave tile 128x32 = 8x2 WMMA tiles
    const int wm0 = (wid & 1) * 128;
    const int wn0 = (wid >> 1) * 32;

    // ---- A fill: one thread per tile row, 128B (8 x b128) per row ----
    const int aRow = tid;
    const _Float16* aSrcBase = x + (size_t)(gm0 + aRow) * GEMM_K;

    // ---- B fill: wave w owns k-rows w*8..w*8+7; lane owns 4 n-columns ----
    const int k0f = wid * 8;
    const int n0f = lane * 4;
    const int8_t* wSrcBase = wq + (size_t)k0f * GEMM_N + gn0 + n0f;

    // per-channel dequant constants for this thread's 4 fixed columns
    half2v s2[4], c2[4];
    {
        const _Float16* sp = scales + gn0 + n0f;
#pragma unroll
        for (int j = 0; j < 4; ++j) {
            _Float16 s = sp[j];
            _Float16 c = (_Float16)(-1152.0f * (float)s);
            s2[j] = (half2v){s, s};
            c2[j] = (half2v){c, c};
        }
    }

    v8f acc[8][2];
#pragma unroll
    for (int mt = 0; mt < 8; ++mt)
#pragma unroll
        for (int nt = 0; nt < 2; ++nt)
            acc[mt][nt] = (v8f){0.f, 0.f, 0.f, 0.f, 0.f, 0.f, 0.f, 0.f};

    for (int kt = 0; kt < GEMM_K; kt += BK) {
        // ---------------- stage A tile ----------------
#if HAVE_ASYNC_LDS
#pragma unroll
        for (int i = 0; i < 8; ++i)
            __builtin_amdgcn_global_load_async_to_lds_b128(
                (AS1 v4i*)(aSrcBase + kt + i * 8),
                (AS3 v4i*)(uint32_t)(uintptr_t)&Als[aRow][i * 8],
                0, 0);
#else
        {
            const uint4* as = (const uint4*)(aSrcBase + kt);
            uint4* ad = (uint4*)&Als[aRow][0];
#pragma unroll
            for (int i = 0; i < 8; ++i) ad[i] = as[i];
        }
#endif
        // ---------------- stage B tile (coalesced b32, dequant, [N][K]) ----
        {
            const int8_t* wp = wSrcBase + (size_t)kt * GEMM_N;
            unsigned q[8];
#pragma unroll
            for (int kk = 0; kk < 8; ++kk)
                q[kk] = *(const unsigned*)(wp + (size_t)kk * GEMM_N);
#pragma unroll
            for (unsigned j = 0; j < 4; ++j) {
                uint4 v;
                v.x = dq2(q[1], q[0], j, s2[j], c2[j]);
                v.y = dq2(q[3], q[2], j, s2[j], c2[j]);
                v.z = dq2(q[5], q[4], j, s2[j], c2[j]);
                v.w = dq2(q[7], q[6], j, s2[j], c2[j]);
                *(uint4*)&Bls[n0f + j][k0f] = v;
            }
        }
        // prefetch next K-tile into L2 while this one is consumed
        if (kt + BK < GEMM_K) {
            __builtin_prefetch(aSrcBase + kt + BK, 0, 0);
            __builtin_prefetch(wSrcBase + (size_t)(kt + BK) * GEMM_N, 0, 0);
        }
#if HAVE_ASYNC_LDS
#if __has_builtin(__builtin_amdgcn_s_wait_asynccnt)
        __builtin_amdgcn_s_wait_asynccnt(0);
#else
        asm volatile("s_wait_asynccnt 0" ::: "memory");
#endif
#endif
        __syncthreads();

        // ---------------- compute: 2 k-steps of 32 ----------------
#pragma unroll
        for (int s = 0; s < 2; ++s) {
            const int k0 = s * 32;
            AFrag bf[2];
#pragma unroll
            for (int nt = 0; nt < 2; ++nt) {
                const int n  = wn0 + nt * 16 + ln15;
                const int kb = k0 + lhi * 16;      // lanes16-31 hold K 16..31
                bf[nt].q[0] = *(const uint4*)&Bls[n][kb];
                bf[nt].q[1] = *(const uint4*)&Bls[n][kb + 8];
            }
            AFrag af[8];
#pragma unroll
            for (int mt = 0; mt < 8; ++mt) {
                const int row = wm0 + mt * 16 + ln15;
                const int kb  = k0 + lhi * 16;     // lanes16-31 hold K 16..31
                af[mt].q[0] = *(const uint4*)&Als[row][kb];
                af[mt].q[1] = *(const uint4*)&Als[row][kb + 8];
            }
#pragma unroll
            for (int mt = 0; mt < 8; ++mt)
#pragma unroll
                for (int nt = 0; nt < 2; ++nt)
                    acc[mt][nt] = __builtin_amdgcn_wmma_f32_16x16x32_f16(
                        false, af[mt].h, false, bf[nt].h,
                        (short)0, acc[mt][nt], false, false);
        }
        __syncthreads();
    }

    // ---------------- epilogue: + bias, fp16 store ----------------
#pragma unroll
    for (int nt = 0; nt < 2; ++nt) {
        const int ncol = gn0 + wn0 + nt * 16 + ln15;
        const float bv = (float)bias[ncol];
#pragma unroll
        for (int mt = 0; mt < 8; ++mt) {
            const int mrow0 = gm0 + wm0 + mt * 16 + lhi * 8;
#pragma unroll
            for (int r = 0; r < 8; ++r) {
                out[(size_t)(mrow0 + r) * GEMM_N + ncol] =
                    (_Float16)(acc[mt][nt][r] + bv);
            }
        }
    }
}

extern "C" void kernel_launch(void* const* d_in, const int* in_sizes, int n_in,
                              void* d_out, int out_size, void* d_ws, size_t ws_size,
                              hipStream_t stream) {
    (void)in_sizes; (void)n_in; (void)out_size; (void)d_ws; (void)ws_size;
    const _Float16* x  = (const _Float16*)d_in[0];
    const int8_t*   w  = (const int8_t*)  d_in[1];
    const _Float16* sc = (const _Float16*)d_in[2];
    const _Float16* bi = (const _Float16*)d_in[3];
    _Float16*       o  = (_Float16*)d_out;

    dim3 grid(GEMM_N / BN, GEMM_M / BM);   // (128, 32)
    dim3 block(256);
    w8a16_gemm_wmma<<<grid, block, 0, stream>>>(x, w, sc, bi, o);
}
